// ImprovedGNNModel_29643864277619
// MI455X (gfx1250) — matrix operand order repaired and checked by reading, not compiled, not run on hardware
//
#include <hip/hip_runtime.h>
#include <hip/hip_bf16.h>
#include <math.h>

// ============================================================================
// ImprovedGNNModel for MI455X (gfx1250, wave32, WMMA, async-LDS copies)
//
// Node MLP GEMMs run on v_wmma_f32_16x16x32_f16 with f16 operands prepared
// once (activations converted/padded, weights transposed+converted), so the
// GEMM stages tiles as pure 16B copies: GLOBAL_LOAD_ASYNC_TO_LDS_B128 when
// available (double-buffered, overlapped with WMMA), else vector ld/st.
// ============================================================================

typedef __attribute__((ext_vector_type(16))) _Float16 v16h;
typedef __attribute__((ext_vector_type(8)))  _Float16 v8h;
typedef __attribute__((ext_vector_type(8)))  float    v8f;
typedef __attribute__((ext_vector_type(4)))  int      v4i;

#define NN 40000          // nodes
#define NPAD 40064        // nodes padded to 313*128 rows (all tiles in-bounds)
#define NE 320000         // edges
#define NB 200            // graphs
#define HD 128            // hidden dim
#define EPS_GN 1e-5f

#ifndef __has_builtin
#define __has_builtin(x) 0
#endif
#if __has_builtin(__builtin_amdgcn_global_load_async_to_lds_b128)
#define USE_ASYNC_LDS 1
#else
#define USE_ASYNC_LDS 0
#endif

// ---------------------------------------------------------------------------
// Input index map (top-level dict insertion order; params pytree, dict keys
// sorted, lists in order).  See round-0 derivation.
// ---------------------------------------------------------------------------
#define IDX_CONV(i)   (5 + 8*(i))
#define IDX_FCLIN     45
#define IDX_FCNORM    53
#define IDX_LSTM      62
#define IDX_NORM(i)   (74 + 3*(i))

// ========================= elementwise / segment kernels ====================

__global__ void fill_kernel(float* __restrict__ p, float v, int n) {
  for (int i = blockIdx.x * blockDim.x + threadIdx.x; i < n;
       i += gridDim.x * blockDim.x)
    p[i] = v;
}

__global__ void count_kernel(const int* __restrict__ batch, float* __restrict__ cnt, int n) {
  int i = blockIdx.x * blockDim.x + threadIdx.x;
  if (i < n) atomicAdd(&cnt[batch[i]], 1.0f);
}

// z = a + b (f32, in place into a) and f16 copy into dst (layers >= 1, Cin=128)
__global__ void add_cvt_kernel(float* __restrict__ a, const float* __restrict__ b,
                               _Float16* __restrict__ dst, int n) {
  int i = blockIdx.x * blockDim.x + threadIdx.x;
  if (i >= n) return;
  float z = a[i] + b[i];
  a[i] = z;
  dst[i] = (_Float16)z;
}

__global__ void add_inplace_kernel(float* __restrict__ a, const float* __restrict__ b, int n) {
  int i = blockIdx.x * blockDim.x + threadIdx.x;
  if (i < n) a[i] += b[i];
}

// f32 [rows x cin] -> f16 [rows_pad x kpad], zero-pad extra cols/rows
__global__ void cvt_act_kernel(const float* __restrict__ src, _Float16* __restrict__ dst,
                               int rows, int cin, int kpad, int rows_pad) {
  int i = blockIdx.x * blockDim.x + threadIdx.x;
  if (i >= rows_pad * kpad) return;
  int r = i / kpad, c = i - r * kpad;
  float v = (r < rows && c < cin) ? src[(size_t)r * cin + c] : 0.0f;
  dst[i] = (_Float16)v;
}

// W f32 [K x Nc] -> Wt f16 [Nc x kpad] (transposed), zero-pad k >= K
__global__ void cvt_wt_kernel(const float* __restrict__ w, _Float16* __restrict__ wt,
                              int K, int Nc, int kpad) {
  int i = blockIdx.x * blockDim.x + threadIdx.x;
  if (i >= Nc * kpad) return;
  int ncol = i / kpad, k = i - ncol * kpad;
  wt[i] = (_Float16)((k < K) ? w[(size_t)k * Nc + ncol] : 0.0f);
}

// GINE message + scatter: one wave per edge, lanes stride channels.
__global__ void gine_edge_kernel(const float* __restrict__ h,
                                 const int* __restrict__ ei,
                                 const float* __restrict__ ea,
                                 const float* __restrict__ We,
                                 const float* __restrict__ be,
                                 float* __restrict__ agg,
                                 int E, int Cin) {
  int e    = blockIdx.x * 8 + (threadIdx.x >> 5);   // 8 waves / block (wave32)
  int lane = threadIdx.x & 31;
  if (e >= E) return;
  int src = ei[e];
  int dst = ei[E + e];
  float a0 = ea[e * 4 + 0], a1 = ea[e * 4 + 1];
  float a2 = ea[e * 4 + 2], a3 = ea[e * 4 + 3];
  for (int c = lane; c < Cin; c += 32) {
    float ev = be[c] + a0 * We[c] + a1 * We[Cin + c] +
               a2 * We[2 * Cin + c] + a3 * We[3 * Cin + c];
    float m = h[(size_t)src * Cin + c] + ev;
    m = fmaxf(m, 0.0f);
    atomicAdd(&agg[(size_t)dst * Cin + c], m);
  }
}

// ============================ WMMA GEMM =====================================
// Cf/Ch[M,Nc] = act(A[M,K]h @ WtT + bias).  A: f16 [>=NPAD x K] row-major,
// Wt: f16 [Nc x K] (pre-transposed).  K % 32 == 0, Nc % 128 == 0, all tile
// loads in-bounds by construction.  Block tile 128x128, BK=32, 8 waves.

#define BM 128
#define BN 128
#define BK 32
#define LDW 40   // LDS row stride in halves (32 + 8 pad; 80B keeps 16B align)

__device__ __forceinline__ void async_copy16(const _Float16* g, _Float16* l) {
#if USE_ASYNC_LDS
  v4i* gg = (v4i*)g;   // reinterpret (generic)
  v4i* ll = (v4i*)l;
  __builtin_amdgcn_global_load_async_to_lds_b128(
      (__attribute__((address_space(1))) v4i*)gg,
      (__attribute__((address_space(3))) v4i*)ll, 0, 0);
#else
  *(v8h*)l = *(const v8h*)g;
#endif
}

__device__ __forceinline__ void stage_tile(const _Float16* __restrict__ A,
                                           const _Float16* __restrict__ Wt,
                                           _Float16* as, _Float16* bs,
                                           int m0, int n0, int K, int k0, int tid) {
#pragma unroll
  for (int i = 0; i < 2; ++i) {
    int c   = tid + i * 256;      // chunk 0..511 (8 halves each)
    int row = c >> 2;
    int seg = (c & 3) * 8;
    async_copy16(A  + (size_t)(m0 + row) * K + k0 + seg, as + row * LDW + seg);
    async_copy16(Wt + (size_t)(n0 + row) * K + k0 + seg, bs + row * LDW + seg);
  }
}

__device__ __forceinline__ void async_wait_all() {
#if USE_ASYNC_LDS
#if __has_builtin(__builtin_amdgcn_s_wait_asynccnt)
  __builtin_amdgcn_s_wait_asynccnt(0);
#else
  asm volatile("s_wait_asynccnt 0" ::: "memory");
#endif
#endif
}

__global__ __launch_bounds__(256) void wmma_gemm_bias_relu(
    const _Float16* __restrict__ A, const _Float16* __restrict__ Wt,
    const float* __restrict__ bias, float* __restrict__ Cf,
    _Float16* __restrict__ Ch, int M, int K, int Nc, int relu) {
  __shared__ __attribute__((aligned(16))) _Float16 As[2][BM * LDW];
  __shared__ __attribute__((aligned(16))) _Float16 Bs[2][BN * LDW];

  const int tid  = threadIdx.x;
  const int lane = tid & 31;
  const int wid  = tid >> 5;          // 0..7
  const int wm0  = (wid >> 2) * 64;   // wave M origin in tile
  const int wn0  = (wid & 3) * 32;    // wave N origin in tile
  const int m0   = blockIdx.y * BM;
  const int n0   = blockIdx.x * BN;

  v8f acc[4][2];
  {
    v8f z = {};
#pragma unroll
    for (int i = 0; i < 4; i++)
#pragma unroll
      for (int j = 0; j < 2; j++) acc[i][j] = z;
  }

  const int T = K / BK;
  stage_tile(A, Wt, As[0], Bs[0], m0, n0, K, 0, tid);

  for (int t = 0; t < T; ++t) {
    async_wait_all();        // this wave's copies for tile t have landed
    __syncthreads();         // all waves' copies landed; prev reads retired
    if (t + 1 < T)           // overlap next copy with this tile's WMMAs
      stage_tile(A, Wt, As[(t + 1) & 1], Bs[(t + 1) & 1],
                 m0, n0, K, (t + 1) * BK, tid);

    const _Float16* as = As[t & 1];
    const _Float16* bs = Bs[t & 1];

    // CDNA5 16-bit WMMA fragment layouts:
    // A (16x32): lanes 0-15 row=lane hold K{0..7,16..23};
    //            lanes 16-31 row=lane-16 hold K{8..15,24..31}.
    // B (32x16): lane holds col n=lane&15; K{0..7,16..23} (lanes<16) or K{8..15,24..31}.
    v16h afrag[4], bfrag[2];
    {
      int arow = lane & 15;
      int klo  = (lane < 16) ? 0 : 8;
#pragma unroll
      for (int mt = 0; mt < 4; mt++) {
        const _Float16* p = &as[(wm0 + mt * 16 + arow) * LDW + klo];
        union { v16h v; v8h h[2]; } u;
        u.h[0] = *(const v8h*)p;
        u.h[1] = *(const v8h*)(p + 16);
        afrag[mt] = u.v;
      }
      int bcol = lane & 15;
      int kb   = (lane < 16) ? 0 : 8;
#pragma unroll
      for (int nt = 0; nt < 2; nt++) {
        const _Float16* p = &bs[(wn0 + nt * 16 + bcol) * LDW + kb];
        union { v16h v; v8h h[2]; } u;
        u.h[0] = *(const v8h*)p;
        u.h[1] = *(const v8h*)(p + 16);
        bfrag[nt] = u.v;
      }
    }
#pragma unroll
    for (int mt = 0; mt < 4; mt++)
#pragma unroll
      for (int nt = 0; nt < 2; nt++)
        acc[mt][nt] = __builtin_amdgcn_wmma_f32_16x16x32_f16(
            false, afrag[mt], false, bfrag[nt], (short)0, acc[mt][nt],
            false, false);
    __syncthreads();
  }

  // Epilogue: C/D layout -> lanes 0-15: M=r, lanes 16-31: M=r+8, N=lane&15.
  const int hi = lane >> 4;
  const int cn = lane & 15;
#pragma unroll
  for (int mt = 0; mt < 4; mt++) {
#pragma unroll
    for (int nt = 0; nt < 2; nt++) {
      int col = n0 + wn0 + nt * 16 + cn;
      float bv = bias[col];
#pragma unroll
      for (int r = 0; r < 8; r++) {
        int row = m0 + wm0 + mt * 16 + hi * 8 + r;
        if (row < M) {
          float v = acc[mt][nt][r] + bv;
          if (relu) v = fmaxf(v, 0.0f);
          if (Cf) Cf[(size_t)row * Nc + col] = v;
          if (Ch) Ch[(size_t)row * Nc + col] = (_Float16)v;
        }
      }
    }
  }
}

// ============================ GraphNorm (batched) ===========================

__global__ void gn_sum_kernel(const float* __restrict__ z, const int* __restrict__ batch,
                              float* __restrict__ gsum, int n) {
  int i = blockIdx.x * blockDim.x + threadIdx.x;
  if (i >= n * HD) return;
  int node = i >> 7, c = i & 127;
  atomicAdd(&gsum[batch[node] * HD + c], z[i]);
}

__global__ void gn_mean_kernel(float* __restrict__ gsum, const float* __restrict__ cnt, int nb) {
  int i = blockIdx.x * blockDim.x + threadIdx.x;
  if (i >= nb * HD) return;
  gsum[i] /= fmaxf(cnt[i >> 7], 1.0f);
}

__global__ void gn_center_kernel(const float* __restrict__ z, const float* __restrict__ mean,
                                 const float* __restrict__ ms, const int* __restrict__ batch,
                                 float* __restrict__ outz, float* __restrict__ gvar, int n) {
  int i = blockIdx.x * blockDim.x + threadIdx.x;
  if (i >= n * HD) return;
  int node = i >> 7, c = i & 127;
  int b = batch[node];
  float o = z[i] - mean[b * HD + c] * ms[c];
  outz[i] = o;
  atomicAdd(&gvar[b * HD + c], o * o);
}

__global__ void gn_final_kernel(const float* __restrict__ outz, const float* __restrict__ gvar,
                                const float* __restrict__ cnt, const float* __restrict__ w,
                                const float* __restrict__ bia, const int* __restrict__ batch,
                                float* __restrict__ h, int n, int residual) {
  int i = blockIdx.x * blockDim.x + threadIdx.x;
  if (i >= n * HD) return;
  int node = i >> 7, c = i & 127;
  int b = batch[node];
  float var = gvar[b * HD + c] / fmaxf(cnt[b], 1.0f);
  float y = w[c] * outz[i] / sqrtf(var + EPS_GN) + bia[c];
  y = fmaxf(y, 0.0f);
  h[i] = residual ? (h[i] + y) : y;
}

// ============================ Set2Set =======================================

__global__ void lstm_cell_kernel(const float* __restrict__ inp, const float* __restrict__ h_old,
                                 const float* __restrict__ c_old, float* __restrict__ h_new,
                                 float* __restrict__ c_new,
                                 const float* __restrict__ w_ih, const float* __restrict__ w_hh,
                                 const float* __restrict__ b_ih, const float* __restrict__ b_hh,
                                 int nb, int d) {
  int idx = blockIdx.x * blockDim.x + threadIdx.x;
  if (idx >= nb * HD) return;
  int b = idx >> 7, c = idx & 127;
  float g[4];
#pragma unroll
  for (int gi = 0; gi < 4; gi++) {
    int row = gi * HD + c;
    float s = b_ih[row] + b_hh[row];
    const float* wi = w_ih + (size_t)row * d;
    const float* xv = inp + (size_t)b * d;
    for (int k = 0; k < d; k++) s += xv[k] * wi[k];
    const float* wh = w_hh + (size_t)row * HD;
    const float* hv = h_old + (size_t)b * HD;
    for (int k = 0; k < HD; k++) s += hv[k] * wh[k];
    g[gi] = s;
  }
  float ii = 1.0f / (1.0f + expf(-g[0]));
  float ff = 1.0f / (1.0f + expf(-g[1]));
  float gg = tanhf(g[2]);
  float oo = 1.0f / (1.0f + expf(-g[3]));
  float cn = ff * c_old[idx] + ii * gg;
  c_new[idx] = cn;
  h_new[idx] = oo * tanhf(cn);
}

__global__ void attn_e_kernel(const float* __restrict__ h, const float* __restrict__ q,
                              const int* __restrict__ batch, float* __restrict__ e, int n) {
  int node = blockIdx.x * 8 + (threadIdx.x >> 5);
  int lane = threadIdx.x & 31;
  if (node >= n) return;
  int b = batch[node];
  float s = 0.0f;
  for (int c = lane; c < HD; c += 32)
    s += h[(size_t)node * HD + c] * q[b * HD + c];
  for (int off = 16; off > 0; off >>= 1) s += __shfl_xor(s, off, 32);
  if (lane == 0) e[node] = s;
}

__device__ inline void atomicMaxF(float* addr, float v) {
  if (v >= 0.0f) atomicMax((int*)addr, __float_as_int(v));
  else atomicMin((unsigned int*)addr, __float_as_uint(v));
}

__global__ void attn_max_kernel(const float* __restrict__ e, const int* __restrict__ batch,
                                float* __restrict__ m, int n) {
  int i = blockIdx.x * blockDim.x + threadIdx.x;
  if (i < n) atomicMaxF(&m[batch[i]], e[i]);
}

__global__ void attn_expsum_kernel(float* __restrict__ e, const float* __restrict__ m,
                                   const int* __restrict__ batch, float* __restrict__ den, int n) {
  int i = blockIdx.x * blockDim.x + threadIdx.x;
  if (i >= n) return;
  int b = batch[i];
  float w = expf(e[i] - m[b]);
  e[i] = w;
  atomicAdd(&den[b], w);
}

__global__ void attn_r_kernel(const float* __restrict__ w, const float* __restrict__ den,
                              const float* __restrict__ h, const int* __restrict__ batch,
                              float* __restrict__ r, int n) {
  int node = blockIdx.x;
  int c    = threadIdx.x;
  if (node >= n) return;
  int b = batch[node];
  float a = w[node] / (den[b] + 1e-16f);
  atomicAdd(&r[b * HD + c], a * h[(size_t)node * HD + c]);
}

__global__ void qstar_kernel(const float* __restrict__ q, const float* __restrict__ r,
                             float* __restrict__ qs, int nb) {
  int i = blockIdx.x * blockDim.x + threadIdx.x;
  if (i >= nb * 256) return;
  int b = i >> 8, c = i & 255;
  qs[i] = (c < HD) ? q[b * HD + c] : r[b * HD + (c - HD)];
}

// ============================ FC head =======================================

__global__ void concat_head_kernel(const float* __restrict__ qs, const float* __restrict__ gf,
                                   float* __restrict__ g0, int nb) {
  int i = blockIdx.x * blockDim.x + threadIdx.x;
  if (i >= nb * 265) return;
  int b = i / 265, c = i % 265;
  g0[i] = (c < 256) ? qs[b * 256 + c] : gf[b * 9 + (c - 256)];
}

__global__ void small_gemm_kernel(const float* __restrict__ A, const float* __restrict__ W,
                                  const float* __restrict__ bia, float* __restrict__ C,
                                  int M, int K, int Nc) {
  int i = blockIdx.x * blockDim.x + threadIdx.x;
  if (i >= M * Nc) return;
  int m = i / Nc, ncol = i - m * Nc;
  float s = bia[ncol];
  const float* a = A + (size_t)m * K;
  for (int k = 0; k < K; k++) s += a[k] * W[(size_t)k * Nc + ncol];
  C[i] = s;
}

// GraphNorm with batch=None (all rows one graph): one block per channel.
__global__ void dense_gn_kernel(const float* __restrict__ g, const float* __restrict__ w,
                                const float* __restrict__ bia, const float* __restrict__ ms,
                                float* __restrict__ out, int rows, int C) {
  int c = blockIdx.x;
  int t = threadIdx.x;
  __shared__ float red[256];
  float s = 0.0f;
  for (int r = t; r < rows; r += 256) s += g[(size_t)r * C + c];
  red[t] = s; __syncthreads();
  for (int o = 128; o > 0; o >>= 1) { if (t < o) red[t] += red[t + o]; __syncthreads(); }
  float mean = red[0] / rows;
  __syncthreads();
  float vs = 0.0f;
  for (int r = t; r < rows; r += 256) {
    float o = g[(size_t)r * C + c] - mean * ms[c];
    vs += o * o;
  }
  red[t] = vs; __syncthreads();
  for (int o = 128; o > 0; o >>= 1) { if (t < o) red[t] += red[t + o]; __syncthreads(); }
  float inv = 1.0f / sqrtf(red[0] / rows + EPS_GN);
  for (int r = t; r < rows; r += 256) {
    float o = g[(size_t)r * C + c] - mean * ms[c];
    out[(size_t)r * C + c] = fmaxf(w[c] * o * inv + bia[c], 0.0f);
  }
}

__global__ void final_lin_kernel(const float* __restrict__ g, const float* __restrict__ w,
                                 const float* __restrict__ bia, float* __restrict__ out, int nb) {
  int i = blockIdx.x * blockDim.x + threadIdx.x;
  if (i >= nb) return;
  float s = bia[0];
  for (int k = 0; k < HD; k++) s += g[(size_t)i * HD + k] * w[k];
  out[i] = s;
}

// ============================ launcher ======================================

extern "C" void kernel_launch(void* const* d_in, const int* in_sizes, int n_in,
                              void* d_out, int out_size, void* d_ws, size_t ws_size,
                              hipStream_t stream) {
  (void)in_sizes; (void)n_in; (void)out_size; (void)ws_size;

  const float* x     = (const float*)d_in[0];
  const int*   ei    = (const int*)d_in[1];
  const float* ea    = (const float*)d_in[2];
  const int*   batch = (const int*)d_in[3];
  const float* gf    = (const float*)d_in[4];
  float* out = (float*)d_out;

  // ---- workspace carve (16B-aligned slices) ----
  float* ws = (float*)d_ws;
  size_t off = 0;
  auto carve = [&](size_t nfloats) {
    float* p = ws + off;
    off += (nfloats + 3) & ~(size_t)3;
    return p;
  };
  auto carveh = [&](size_t nhalf) { return (_Float16*)carve((nhalf + 1) / 2); };

  float* Wh   = carve((size_t)NN * HD);     // node features (f32)
  float* Wagg = carve((size_t)NN * HD);     // edge aggregate / z0 (f32)
  float* Wt2  = carve((size_t)NN * HD);     // centered-z scratch for GraphNorm
  float* Wz   = carve((size_t)NN * HD);     // MLP output (f32)
  float* Wcnt = carve(NB);
  float* Wgs  = carve((size_t)NB * HD);
  float* Wgv  = carve((size_t)NB * HD);
  float* Wqs  = carve((size_t)NB * 256);
  float* Whs0 = carve((size_t)3 * NB * HD);
  float* Whs1 = carve((size_t)3 * NB * HD);
  float* Wcs0 = carve((size_t)3 * NB * HD);
  float* Wcs1 = carve((size_t)3 * NB * HD);
  float* Wev  = carve(NN);
  float* Wm   = carve(NB);
  float* Wden = carve(NB);
  float* Wr   = carve((size_t)NB * HD);
  float* Wg0  = carve((size_t)NB * 265);
  float* Wf1  = carve((size_t)NB * 512);
  float* Wn1  = carve((size_t)NB * 512);
  float* Wf2  = carve((size_t)NB * 256);
  float* Wn2  = carve((size_t)NB * 256);
  float* Wf3  = carve((size_t)NB * HD);
  float* Wn3  = carve((size_t)NB * HD);
  // f16 buffers (padded to NPAD rows, K padded to multiple of 32)
  _Float16* Xh   = carveh((size_t)NPAD * HD);    // GEMM1 input
  _Float16* T1h  = carveh((size_t)NPAD * 256);   // GEMM1 out / GEMM2 in
  _Float16* T2h  = carveh((size_t)NPAD * HD);    // GEMM2 out / GEMM3 in
  _Float16* Wt1h = carveh((size_t)256 * 128);    // [256 x Kpad], Kpad<=128
  _Float16* Wt2h = carveh((size_t)128 * 256);    // [128 x 256]
  _Float16* Wt3h = carveh((size_t)128 * 128);    // [128 x 128]

  auto fill = [&](float* p, float v, int n) {
    fill_kernel<<<dim3(1024), dim3(256), 0, stream>>>(p, v, n);
  };
  auto gemm = [&](const _Float16* A, const _Float16* Wt, const float* b,
                  float* Cf, _Float16* Ch, int K, int Nc, int relu) {
    dim3 g(Nc / BN, NPAD / BM);
    wmma_gemm_bias_relu<<<g, 256, 0, stream>>>(A, Wt, b, Cf, Ch, NN, K, Nc, relu);
  };
  const int TPB = 256;
  auto blocks = [](long n) { return (unsigned)((n + 255) / 256); };

  // ---- per-graph node counts ----
  fill(Wcnt, 0.0f, NB);
  count_kernel<<<blocks(NN), TPB, 0, stream>>>(batch, Wcnt, NN);

  // ---- 5 GINE layers ----
  for (int i = 0; i < 5; i++) {
    int Cin  = (i == 0) ? 16 : HD;
    int Kp   = (i == 0) ? 32 : HD;   // K padded to 32-multiple
    const float* h_in = (i == 0) ? x : Wh;
    const float* elb = (const float*)d_in[IDX_CONV(i) + 0];
    const float* elw = (const float*)d_in[IDX_CONV(i) + 1];
    const float* b1  = (const float*)d_in[IDX_CONV(i) + 2];
    const float* w1  = (const float*)d_in[IDX_CONV(i) + 3];
    const float* b2  = (const float*)d_in[IDX_CONV(i) + 4];
    const float* w2  = (const float*)d_in[IDX_CONV(i) + 5];
    const float* b3  = (const float*)d_in[IDX_CONV(i) + 6];
    const float* w3  = (const float*)d_in[IDX_CONV(i) + 7];

    // weights: f32 [K x Nc] -> f16 [Nc x Kpad] transposed
    cvt_wt_kernel<<<blocks(256L * Kp), TPB, 0, stream>>>(w1, Wt1h, Cin, 256, Kp);
    cvt_wt_kernel<<<blocks(128L * 256), TPB, 0, stream>>>(w2, Wt2h, 256, 128, 256);
    cvt_wt_kernel<<<blocks(128L * 128), TPB, 0, stream>>>(w3, Wt3h, 128, 128, 128);

    fill(Wagg, 0.0f, NN * Cin);
    gine_edge_kernel<<<(NE + 7) / 8, TPB, 0, stream>>>(h_in, ei, ea, elw, elb,
                                                       Wagg, NE, Cin);
    if (i == 0) {
      add_inplace_kernel<<<blocks((long)NN * Cin), TPB, 0, stream>>>(
          Wagg, h_in, NN * Cin);
      cvt_act_kernel<<<blocks((long)NPAD * Kp), TPB, 0, stream>>>(
          Wagg, Xh, NN, Cin, Kp, NPAD);
    } else {
      add_cvt_kernel<<<blocks((long)NN * HD), TPB, 0, stream>>>(
          Wagg, h_in, Xh, NN * HD);
    }

    gemm(Xh,  Wt1h, b1, nullptr, T1h, Kp,  256, 1);  // relu(z @ W1 + b1) -> f16
    gemm(T1h, Wt2h, b2, nullptr, T2h, 256, HD,  1);  // relu(.. @ W2 + b2) -> f16
    gemm(T2h, Wt3h, b3, Wz, nullptr,  HD,  HD,  0);  // .. @ W3 + b3 -> f32

    const float* gnb = (const float*)d_in[IDX_NORM(i) + 0];
    const float* gnm = (const float*)d_in[IDX_NORM(i) + 1];
    const float* gnw = (const float*)d_in[IDX_NORM(i) + 2];
    fill(Wgs, 0.0f, NB * HD);
    fill(Wgv, 0.0f, NB * HD);
    gn_sum_kernel<<<blocks((long)NN * HD), TPB, 0, stream>>>(Wz, batch, Wgs, NN);
    gn_mean_kernel<<<blocks((long)NB * HD), TPB, 0, stream>>>(Wgs, Wcnt, NB);
    gn_center_kernel<<<blocks((long)NN * HD), TPB, 0, stream>>>(
        Wz, Wgs, gnm, batch, Wt2, Wgv, NN);
    gn_final_kernel<<<blocks((long)NN * HD), TPB, 0, stream>>>(
        Wt2, Wgv, Wcnt, gnw, gnb, batch, Wh, NN, (i > 0) ? 1 : 0);
  }

  // ---- Set2Set (4 steps, 3-layer LSTM, B=200) ----
  fill(Wqs, 0.0f, NB * 256);
  fill(Whs0, 0.0f, 3 * NB * HD);
  fill(Wcs0, 0.0f, 3 * NB * HD);
  float *hsA = Whs0, *hsB = Whs1, *csA = Wcs0, *csB = Wcs1;
  for (int step = 0; step < 4; step++) {
    const float* inp = Wqs;
    int d = 256;
    for (int l = 0; l < 3; l++) {
      const float* bhh = (const float*)d_in[IDX_LSTM + 4 * l + 0];
      const float* bih = (const float*)d_in[IDX_LSTM + 4 * l + 1];
      const float* whh = (const float*)d_in[IDX_LSTM + 4 * l + 2];
      const float* wih = (const float*)d_in[IDX_LSTM + 4 * l + 3];
      lstm_cell_kernel<<<blocks((long)NB * HD), TPB, 0, stream>>>(
          inp, hsA + (size_t)l * NB * HD, csA + (size_t)l * NB * HD,
          hsB + (size_t)l * NB * HD, csB + (size_t)l * NB * HD,
          wih, whh, bih, bhh, NB, d);
      inp = hsB + (size_t)l * NB * HD;
      d = HD;
    }
    const float* q = inp;  // [B,128]
    attn_e_kernel<<<(NN + 7) / 8, TPB, 0, stream>>>(Wh, q, batch, Wev, NN);
    fill(Wm, -3.0e38f, NB);
    attn_max_kernel<<<blocks(NN), TPB, 0, stream>>>(Wev, batch, Wm, NN);
    fill(Wden, 0.0f, NB);
    attn_expsum_kernel<<<blocks(NN), TPB, 0, stream>>>(Wev, Wm, batch, Wden, NN);
    fill(Wr, 0.0f, NB * HD);
    attn_r_kernel<<<NN, HD, 0, stream>>>(Wev, Wden, Wh, batch, Wr, NN);
    qstar_kernel<<<blocks((long)NB * 256), TPB, 0, stream>>>(q, Wr, Wqs, NB);
    float* t;
    t = hsA; hsA = hsB; hsB = t;
    t = csA; csA = csB; csB = t;
  }

  // ---- FC head ----
  const float* f0b = (const float*)d_in[IDX_FCLIN + 0];
  const float* f0w = (const float*)d_in[IDX_FCLIN + 1];
  const float* f1b = (const float*)d_in[IDX_FCLIN + 2];
  const float* f1w = (const float*)d_in[IDX_FCLIN + 3];
  const float* f2b = (const float*)d_in[IDX_FCLIN + 4];
  const float* f2w = (const float*)d_in[IDX_FCLIN + 5];
  const float* f3b = (const float*)d_in[IDX_FCLIN + 6];
  const float* f3w = (const float*)d_in[IDX_FCLIN + 7];

  concat_head_kernel<<<blocks((long)NB * 265), TPB, 0, stream>>>(Wqs, gf, Wg0, NB);

  small_gemm_kernel<<<blocks((long)NB * 512), TPB, 0, stream>>>(Wg0, f0w, f0b, Wf1, NB, 265, 512);
  dense_gn_kernel<<<512, 256, 0, stream>>>(
      Wf1, (const float*)d_in[IDX_FCNORM + 2], (const float*)d_in[IDX_FCNORM + 0],
      (const float*)d_in[IDX_FCNORM + 1], Wn1, NB, 512);

  small_gemm_kernel<<<blocks((long)NB * 256), TPB, 0, stream>>>(Wn1, f1w, f1b, Wf2, NB, 512, 256);
  dense_gn_kernel<<<256, 256, 0, stream>>>(
      Wf2, (const float*)d_in[IDX_FCNORM + 5], (const float*)d_in[IDX_FCNORM + 3],
      (const float*)d_in[IDX_FCNORM + 4], Wn2, NB, 256);

  small_gemm_kernel<<<blocks((long)NB * HD), TPB, 0, stream>>>(Wn2, f2w, f2b, Wf3, NB, 256, HD);
  dense_gn_kernel<<<HD, 256, 0, stream>>>(
      Wf3, (const float*)d_in[IDX_FCNORM + 8], (const float*)d_in[IDX_FCNORM + 6],
      (const float*)d_in[IDX_FCNORM + 7], Wn3, NB, HD);

  final_lin_kernel<<<blocks(NB), TPB, 0, stream>>>(Wn3, f3w, f3b, out, NB);
}
